// Model_30657476559127
// MI455X (gfx1250) — compile-verified, matrix operand order checked by reading, not run on hardware
//
#include <hip/hip_runtime.h>
#include <stdint.h>
#include <stddef.h>

// ---------------------------------------------------------------------------
// Types for CDNA5 WMMA (wave32, 16x16x32 f16 -> f32)
// ---------------------------------------------------------------------------
typedef _Float16 half_t;
typedef __attribute__((ext_vector_type(16))) _Float16 v16h;
typedef __attribute__((ext_vector_type(8)))  _Float16 v8h;
typedef __attribute__((ext_vector_type(8)))  float    v8f;

#define BATCH   8
#define NPOINTS 8192
#define NCOARSE 384
#define NFINE   6144

struct Lay { const float *W, *g, *b; int cin, cout; };

static __device__ __forceinline__ v16h load_frag(const half_t* p0, const half_t* p1) {
  v8h lo = *(const v8h*)p0;
  v8h hi = *(const v8h*)p1;
  v16h r;
#pragma unroll
  for (int i = 0; i < 8; ++i) { r[i] = lo[i]; r[i + 8] = hi[i]; }
  return r;
}

// ---------------------------------------------------------------------------
// WMMA GEMM: Y[Mpad x Npad] = X[Mpad x Kpad] * W[Npad x Kpad]^T (+bias)(+relu)
// One wave owns a 16(M) x NT*16(N) strip; A fragment reused across NT tiles.
// Per-lane addressing follows the ISA 16-bit A-matrix 16x32 layout:
//   lane<16 : row = r, K in [k..k+7] U [k+16..k+23]
//   lane>=16: row = r, K in [k+8..k+15] U [k+24..k+31]
// B mirrors A with the tile column (row of W) in place of M.
// K loop is software-pipelined with two fragment sets (unroll-by-2) so loads
// for chunk c+1 are in flight while WMMAs for chunk c execute.
// ---------------------------------------------------------------------------
template <int NT>
static __device__ __forceinline__ void gemm_strip(
    const half_t* __restrict__ X, const half_t* __restrict__ Wm,
    const float* __restrict__ bias, float* __restrict__ Y,
    int m0, int n0, int Kpad, int Npad, int Nreal, int relu)
{
  int lane = threadIdx.x & 31;
  int hh   = lane >> 4;      // which K-half this lane holds
  int r    = lane & 15;      // row (A) / col (B) within tile

  v8f acc[NT];
#pragma unroll
  for (int t = 0; t < NT; ++t)
#pragma unroll
    for (int i = 0; i < 8; ++i) acc[t][i] = 0.0f;

  const half_t* xrow = X + (size_t)(m0 + r) * Kpad + hh * 8;
  const half_t* wrow[NT];
#pragma unroll
  for (int t = 0; t < NT; ++t)
    wrow[t] = Wm + (size_t)(n0 + t * 16 + r) * Kpad + hh * 8;

  const int nc = Kpad >> 5;          // number of 32-wide K chunks (>=1)

  // fragment set 0 <- chunk 0
  v16h a0 = load_frag(xrow, xrow + 16);
  v16h b0[NT];
#pragma unroll
  for (int t = 0; t < NT; ++t) b0[t] = load_frag(wrow[t], wrow[t] + 16);

  v16h a1;
  v16h b1[NT];
  int c = 1;
  for (; c + 1 < nc; c += 2) {
    int k1 = c << 5;
    a1 = load_frag(xrow + k1, xrow + k1 + 16);
#pragma unroll
    for (int t = 0; t < NT; ++t) b1[t] = load_frag(wrow[t] + k1, wrow[t] + k1 + 16);
#pragma unroll
    for (int t = 0; t < NT; ++t)
      acc[t] = __builtin_amdgcn_wmma_f32_16x16x32_f16(
          false, a0, false, b0[t], (short)0, acc[t], false, false);
    int k2 = (c + 1) << 5;
    a0 = load_frag(xrow + k2, xrow + k2 + 16);
#pragma unroll
    for (int t = 0; t < NT; ++t) b0[t] = load_frag(wrow[t] + k2, wrow[t] + k2 + 16);
#pragma unroll
    for (int t = 0; t < NT; ++t)
      acc[t] = __builtin_amdgcn_wmma_f32_16x16x32_f16(
          false, a1, false, b1[t], (short)0, acc[t], false, false);
  }
  if (c < nc) {                      // one trailing chunk beyond set 0
    int k1 = c << 5;
    a1 = load_frag(xrow + k1, xrow + k1 + 16);
#pragma unroll
    for (int t = 0; t < NT; ++t) b1[t] = load_frag(wrow[t] + k1, wrow[t] + k1 + 16);
#pragma unroll
    for (int t = 0; t < NT; ++t)
      acc[t] = __builtin_amdgcn_wmma_f32_16x16x32_f16(
          false, a0, false, b0[t], (short)0, acc[t], false, false);
#pragma unroll
    for (int t = 0; t < NT; ++t)
      acc[t] = __builtin_amdgcn_wmma_f32_16x16x32_f16(
          false, a1, false, b1[t], (short)0, acc[t], false, false);
  } else {
#pragma unroll
    for (int t = 0; t < NT; ++t)
      acc[t] = __builtin_amdgcn_wmma_f32_16x16x32_f16(
          false, a0, false, b0[t], (short)0, acc[t], false, false);
  }

  // C/D layout: lane half selects rows v (+8); N = r within tile
#pragma unroll
  for (int t = 0; t < NT; ++t) {
    int col = n0 + t * 16 + r;
    float bv = (bias != nullptr && col < Nreal) ? bias[col] : 0.0f;
#pragma unroll
    for (int v = 0; v < 8; ++v) {
      int row = m0 + hh * 8 + v;
      float val = acc[t][v] + bv;
      if (relu) val = fmaxf(val, 0.0f);
      Y[(size_t)row * Npad + col] = val;
    }
  }
}

__global__ void __launch_bounds__(256) k_gemm(
    const half_t* __restrict__ X, const half_t* __restrict__ Wm,
    const float* __restrict__ bias, float* __restrict__ Y,
    int Mpad, int Npad, int Kpad, int Nreal, int relu, int nt64)
{
  // readfirstlane -> tile coordinates live in SGPRs; all edge branches are SALU
  int wave  = __builtin_amdgcn_readfirstlane(blockIdx.x * 8 + (threadIdx.x >> 5));
  int total = (Mpad >> 4) * nt64;
  if (wave >= total) return;
  int tm = wave / nt64, tn = wave % nt64;
  int m0 = tm << 4, n0 = tn << 6;
  int rem = (Npad - n0) >> 4;        // 16-wide tiles remaining in N (>=1)
  if (rem >= 4)      gemm_strip<4>(X, Wm, bias, Y, m0, n0, Kpad, Npad, Nreal, relu);
  else if (rem == 3) gemm_strip<3>(X, Wm, bias, Y, m0, n0, Kpad, Npad, Nreal, relu);
  else if (rem == 2) gemm_strip<2>(X, Wm, bias, Y, m0, n0, Kpad, Npad, Nreal, relu);
  else               gemm_strip<1>(X, Wm, bias, Y, m0, n0, Kpad, Npad, Nreal, relu);
}

// ---------------------------------------------------------------------------
// Weight pack f32 (Nr x K) -> f16 (Npad x Kpad), zero padded
// ---------------------------------------------------------------------------
__global__ void k_pack_weight(const float* __restrict__ W, half_t* __restrict__ dst,
                              int Nr, int K, int Npad, int Kpad)
{
  size_t i = (size_t)blockIdx.x * blockDim.x + threadIdx.x;
  size_t total = (size_t)Npad * Kpad;
  if (i >= total) return;
  int row = (int)(i / Kpad), col = (int)(i % Kpad);
  float v = (row < Nr && col < K) ? W[(size_t)row * K + col] : 0.0f;
  dst[i] = (half_t)v;
}

// ---------------------------------------------------------------------------
// Per-channel batchnorm statistics over M rows (live batch stats)
// scale = g/sqrt(var+eps), shift = b - mean*scale
// ---------------------------------------------------------------------------
__global__ void k_bn_stats(const float* __restrict__ Y, int M, int Npad,
                           const float* __restrict__ gamma, const float* __restrict__ beta,
                           float* __restrict__ scale, float* __restrict__ shift)
{
  __shared__ float s1[256], s2[256];
  int c = blockIdx.x;
  int tid = threadIdx.x;
  float a = 0.f, b = 0.f;
  for (int i = tid; i < M; i += 256) {
    float v = Y[(size_t)i * Npad + c];
    a += v; b += v * v;
  }
  s1[tid] = a; s2[tid] = b;
  __syncthreads();
  for (int off = 128; off > 0; off >>= 1) {
    if (tid < off) { s1[tid] += s1[tid + off]; s2[tid] += s2[tid + off]; }
    __syncthreads();
  }
  if (tid == 0) {
    float mean = s1[0] / (float)M;
    float var  = s2[0] / (float)M - mean * mean;
    float sc   = gamma[c] * rsqrtf(var + 1e-5f);
    scale[c] = sc;
    shift[c] = beta[c] - mean * sc;
  }
}

// affine+relu+convert to f16, zero-padding rows>=M and cols>=Ncols
__global__ void k_pack_act(const float* __restrict__ Y, int ldy, int M, int Mpad,
                           int Ncols, const float* __restrict__ scale,
                           const float* __restrict__ shift, int relu,
                           half_t* __restrict__ dst, int ldd)
{
  size_t i = (size_t)blockIdx.x * blockDim.x + threadIdx.x;
  size_t total = (size_t)Mpad * ldd;
  if (i >= total) return;
  int row = (int)(i / ldd), col = (int)(i % ldd);
  float v = 0.0f;
  if (row < M && col < Ncols) {
    v = Y[(size_t)row * ldy + col];
    if (scale) v = v * scale[col] + shift[col];
    if (relu) v = fmaxf(v, 0.0f);
  }
  dst[i] = (half_t)v;
}

__global__ void k_bn_relu(float* __restrict__ Y, int ldy, int M, int Ncols,
                          const float* __restrict__ scale, const float* __restrict__ shift)
{
  size_t i = (size_t)blockIdx.x * blockDim.x + threadIdx.x;
  size_t total = (size_t)M * Ncols;
  if (i >= total) return;
  int row = (int)(i / Ncols), col = (int)(i % Ncols);
  float v = Y[(size_t)row * ldy + col] * scale[col] + shift[col];
  Y[(size_t)row * ldy + col] = fmaxf(v, 0.0f);
}

// max over groups of gsz consecutive rows
__global__ void k_max_group(const float* __restrict__ Y, int ldy, int G, int gsz,
                            int Ncols, float* __restrict__ out)
{
  size_t i = (size_t)blockIdx.x * blockDim.x + threadIdx.x;
  size_t total = (size_t)G * Ncols;
  if (i >= total) return;
  int g = (int)(i / Ncols), c = (int)(i % Ncols);
  float m = -1e30f;
  for (int s = 0; s < gsz; ++s)
    m = fmaxf(m, Y[((size_t)g * gsz + s) * ldy + c]);
  out[i] = m;
}

// ---------------------------------------------------------------------------
// Farthest point sampling: one workgroup per batch; distance array in LDS
// ---------------------------------------------------------------------------
__global__ void k_fps(const float* __restrict__ pts, int N, int npoint, int* __restrict__ out)
{
  __shared__ float dist[NPOINTS];
  __shared__ float rv[256];
  __shared__ int   ri[256];
  __shared__ float last[3];
  int b = blockIdx.x;
  int tid = threadIdx.x;
  const float* P = pts + (size_t)b * N * 3;
  for (int i = tid; i < N; i += 256) dist[i] = 1e10f;
  if (tid == 0) {
    out[b * npoint] = 0;
    last[0] = P[0]; last[1] = P[1]; last[2] = P[2];
  }
  __syncthreads();
  for (int it = 1; it < npoint; ++it) {
    float lx = last[0], ly = last[1], lz = last[2];
    float bm = -1.0f; int bi = 0;
    for (int i = tid; i < N; i += 256) {
      float dx = P[i * 3 + 0] - lx;
      float dy = P[i * 3 + 1] - ly;
      float dz = P[i * 3 + 2] - lz;
      float d = dx * dx + dy * dy + dz * dz;
      float nd = fminf(dist[i], d);
      dist[i] = nd;
      if (nd > bm) { bm = nd; bi = i; }
    }
    rv[tid] = bm; ri[tid] = bi;
    __syncthreads();
    for (int off = 128; off > 0; off >>= 1) {
      if (tid < off) {
        if (rv[tid + off] > rv[tid] ||
            (rv[tid + off] == rv[tid] && ri[tid + off] < ri[tid])) {
          rv[tid] = rv[tid + off]; ri[tid] = ri[tid + off];
        }
      }
      __syncthreads();
    }
    if (tid == 0) {
      int w = ri[0];
      out[b * npoint + it] = w;
      last[0] = P[w * 3 + 0]; last[1] = P[w * 3 + 1]; last[2] = P[w * 3 + 2];
    }
    __syncthreads();
  }
}

__global__ void k_gather3(const float* __restrict__ pts, const int* __restrict__ idx,
                          float* __restrict__ out, int N, int np)
{
  int i = blockIdx.x * blockDim.x + threadIdx.x;
  int total = BATCH * np * 3;
  if (i >= total) return;
  int c = i % 3, p = (i / 3) % np, b = i / (3 * np);
  out[i] = pts[((size_t)b * N + idx[b * np + p]) * 3 + c];
}

// ball query: first `ns` indices (ascending) within radius; fill with first hit
__global__ void k_ball(const float* __restrict__ xyz, const float* __restrict__ ctr,
                       int* __restrict__ out, int N, int np, float r2, int ns)
{
  int t = blockIdx.x * blockDim.x + threadIdx.x;
  if (t >= BATCH * np) return;
  int b = t / np;
  float cx = ctr[t * 3 + 0], cy = ctr[t * 3 + 1], cz = ctr[t * 3 + 2];
  const float* P = xyz + (size_t)b * N * 3;
  int* o = out + (size_t)t * ns;
  int cnt = 0, first = 0; bool have = false;
  for (int i = 0; i < N && cnt < ns; ++i) {
    float dx = P[i * 3 + 0] - cx;
    float dy = P[i * 3 + 1] - cy;
    float dz = P[i * 3 + 2] - cz;
    if (dx * dx + dy * dy + dz * dz < r2) {
      if (!have) { first = i; have = true; }
      o[cnt++] = i;
    }
  }
  for (int j = cnt; j < ns; ++j) o[j] = first;
}

// grouped rows: [xyz - center | feats | zero-pad] -> f16, width Kpad
__global__ void k_group(const float* __restrict__ xyz, const float* __restrict__ feats,
                        const float* __restrict__ ctr, const int* __restrict__ idx,
                        half_t* __restrict__ dst, int N, int np, int ns, int C, int Kpad)
{
  int t = blockIdx.x * blockDim.x + threadIdx.x;
  int total = BATCH * np * ns;
  if (t >= total) return;
  int s = t % ns;
  int bp = t / ns;                 // b*np + p
  int b = bp / np;
  int gi = idx[(size_t)bp * ns + s];
  const float* pt = xyz + ((size_t)b * N + gi) * 3;
  const float* c0 = ctr + (size_t)bp * 3;
  half_t* o = dst + (size_t)t * Kpad;
  o[0] = (half_t)(pt[0] - c0[0]);
  o[1] = (half_t)(pt[1] - c0[1]);
  o[2] = (half_t)(pt[2] - c0[2]);
  if (feats) {
    const float* fr = feats + ((size_t)b * N + gi) * C;
    for (int c = 0; c < C; ++c) o[3 + c] = (half_t)fr[c];
  }
  for (int c = 3 + C; c < Kpad; ++c) o[c] = (half_t)0.0f;
}

// generic f16 row builder: [a | b | zero-pad], rows>=M zeroed
__global__ void k_rows(half_t* __restrict__ dst, int ldd, int M, int Mpad,
                       const float* __restrict__ a, int ca,
                       const float* __restrict__ b, int cb)
{
  size_t i = (size_t)blockIdx.x * blockDim.x + threadIdx.x;
  size_t total = (size_t)Mpad * ldd;
  if (i >= total) return;
  int row = (int)(i / ldd), col = (int)(i % ldd);
  float v = 0.0f;
  if (row < M) {
    if (col < ca) v = a[(size_t)row * ca + col];
    else if (b != nullptr && col < ca + cb) v = b[(size_t)row * cb + (col - ca)];
  }
  dst[i] = (half_t)v;
}

// decoder feature rows: [gf(1024) | seed(2) | pf(3) | pad] width 1056
__global__ void k_decfeat(const float* __restrict__ gf, const float* __restrict__ cz,
                          half_t* __restrict__ dst)
{
  const int KD = 1056;
  size_t i = (size_t)blockIdx.x * blockDim.x + threadIdx.x;
  size_t total = (size_t)BATCH * NFINE * KD;
  if (i >= total) return;
  int col = (int)(i % KD);
  int row = (int)(i / KD);
  int b = row / NFINE, fN = row % NFINE;
  int cp = fN >> 4, g2 = fN & 15;
  float v = 0.0f;
  if (col < 1024)       v = gf[(size_t)b * 1024 + col];
  else if (col == 1024) v = -0.01f + (0.02f / 3.0f) * (float)(g2 & 3);
  else if (col == 1025) v = -0.01f + (0.02f / 3.0f) * (float)(g2 >> 2);
  else if (col < 1029)  v = cz[(size_t)b * 1152 + cp * 3 + (col - 1026)];
  dst[i] = (half_t)v;
}

// ---------------------------------------------------------------------------
// Output assembly
// ---------------------------------------------------------------------------
__global__ void k_center(float* o) { int i = threadIdx.x; if (i < BATCH * 3) o[i] = 0.0f; }

__global__ void k_fine(const float* __restrict__ Yfc3, const float* __restrict__ cz,
                       float* __restrict__ ofine)
{
  int i = blockIdx.x * blockDim.x + threadIdx.x;
  int total = BATCH * NFINE * 3;
  if (i >= total) return;
  int c = i % 3, bf = i / 3;
  int b = bf / NFINE, fN = bf % NFINE;
  float pf = cz[(size_t)b * 1152 + (fN >> 4) * 3 + c];
  ofine[i] = Yfc3[(size_t)bf * 16 + c] + pf;
}

__global__ void k_pcout(const float* __restrict__ pc, const float* __restrict__ ofine,
                        float* __restrict__ opc)
{
  const int PP = NPOINTS + NFINE;
  int i = blockIdx.x * blockDim.x + threadIdx.x;
  int total = BATCH * PP * 3;
  if (i >= total) return;
  int c = i % 3, bp = i / 3;
  int b = bp / PP, p = bp % PP;
  opc[i] = (p < NPOINTS) ? pc[((size_t)b * NPOINTS + p) * 3 + c]
                         : ofine[((size_t)b * NFINE + (p - NPOINTS)) * 3 + c];
}

__global__ void k_coarseout(const float* __restrict__ cz, const float* __restrict__ pc,
                            const int* __restrict__ idx128, float* __restrict__ oc)
{
  const int CP = NCOARSE + 128;
  int i = blockIdx.x * blockDim.x + threadIdx.x;
  int total = BATCH * CP * 3;
  if (i >= total) return;
  int c = i % 3, bp = i / 3;
  int b = bp / CP, p = bp % CP;
  float v;
  if (p < NCOARSE) v = cz[(size_t)b * 1152 + p * 3 + c];
  else v = pc[((size_t)b * NPOINTS + idx128[b * 128 + (p - NCOARSE)]) * 3 + c];
  oc[i] = v;
}

// ---------------------------------------------------------------------------
// Host orchestration
// ---------------------------------------------------------------------------
// semantic slot -> input index when leaves are flattened alphabetically
static const int MAP_ALPHA[72] = {
  71,
  26,27,28,29,30,31,32,33,34,      // sa1
  35,36,37,38,39,40,41,42,43,      // sa2
  44,45,46,47,48,49,50,51,52,      // sa3
  53,54,55,56,57,58,59,60,61,      // sa4
  62,63,64,65,66,67,68,69,70,      // sa_sep
  4,5, 0,1, 6,7, 2,3, 8,9,         // conv1,bn1,conv2,bn2,conv3
  20,21, 22,23, 24,25,             // mlp1..3
  14,15, 10,11, 16,17, 12,13, 18,19 // fc1,fbn1,fc2,fbn2,fc3
};

extern "C" void kernel_launch(void* const* d_in, const int* in_sizes, int n_in,
                              void* d_out, int out_size, void* d_ws, size_t ws_size,
                              hipStream_t stream) {
  (void)n_in; (void)out_size; (void)ws_size;
  const bool pc_first = (in_sizes[0] == BATCH * NPOINTS * 3);
  auto f = [&](int s) -> const float* {
    int i = pc_first ? s : MAP_ALPHA[s];
    return (const float*)d_in[i];
  };
  const float* pc = f(0);

  char* ws = (char*)d_ws;
  size_t off = 0;
  auto alloc = [&](size_t bytes) -> void* {
    void* p = ws + off;
    off += (bytes + 255) & ~(size_t)255;
    return p;
  };
  auto cdiv = [](size_t a, size_t b) { return (int)((a + b - 1) / b); };
  auto r16 = [](int x) { return (x + 15) & ~15; };
  auto r32 = [](int x) { return (x + 31) & ~31; };

  // big ping-pong buffers
  half_t* bufA = (half_t*)alloc((size_t)49152 * 1056 * sizeof(half_t)); // 104 MB
  half_t* bufB = (half_t*)alloc((size_t)49152 * 512 * sizeof(half_t));  //  50 MB
  float*  bufY = (float*)alloc((size_t)49152 * 512 * sizeof(float));    // 100 MB

  float* d_sc = (float*)alloc(2048 * sizeof(float));
  float* d_sh = (float*)alloc(2048 * sizeof(float));

  float* f1   = (float*)alloc((size_t)4096 * 64 * sizeof(float));
  float* f2   = (float*)alloc((size_t)2048 * 256 * sizeof(float));
  float* f3   = (float*)alloc((size_t)1024 * 512 * sizeof(float));
  float* f4   = (float*)alloc((size_t)8 * 1024 * sizeof(float));
  float* fsep = (float*)alloc((size_t)8 * 1024 * sizeof(float));
  float* gfY  = (float*)alloc((size_t)16 * 1024 * sizeof(float));
  float* czY  = (float*)alloc((size_t)16 * 1152 * sizeof(float));

  int*   idx1   = (int*)alloc(8 * 512 * 4);
  float* x1     = (float*)alloc(8 * 512 * 3 * 4);
  int*   ball1  = (int*)alloc(8 * 512 * 32 * 4);
  int*   idx2   = (int*)alloc(8 * 256 * 4);
  float* x2     = (float*)alloc(8 * 256 * 3 * 4);
  int*   ball2  = (int*)alloc(8 * 256 * 16 * 4);
  int*   idx3   = (int*)alloc(8 * 128 * 4);
  float* x3     = (float*)alloc(8 * 128 * 3 * 4);
  int*   ball3  = (int*)alloc(8 * 128 * 16 * 4);
  int*   idxsep = (int*)alloc(8 * 4 * 4);
  float* xsep   = (float*)alloc(8 * 4 * 3 * 4);
  int*   idx128 = (int*)alloc(8 * 128 * 4);

  auto packW = [&](const float* W, int Nr, int K) -> half_t* {
    int Np = r16(Nr), Kp = r32(K);
    half_t* d = (half_t*)alloc((size_t)Np * Kp * sizeof(half_t));
    size_t tot = (size_t)Np * Kp;
    k_pack_weight<<<cdiv(tot, 256), 256, 0, stream>>>(W, d, Nr, K, Np, Kp);
    return d;
  };

  auto gemm = [&](const half_t* X, const half_t* Wm, const float* bias, float* Y,
                  int Mpad, int N, int K, int relu) {
    int Np = r16(N), Kp = r32(K);
    int nt64 = (Np + 63) / 64;
    int waves = (Mpad / 16) * nt64;
    k_gemm<<<(waves + 7) / 8, 256, 0, stream>>>(X, Wm, bias, Y, Mpad, Np, Kp, N, relu, nt64);
  };

  // shared MLP with live-batch BN + ReLU; final activated result left in bufY
  auto runMLP = [&](half_t* X0, int M, int Mpad, const Lay* L, half_t** Wp) {
    half_t* cur = X0;
    half_t* alt = (X0 == bufA) ? bufB : bufA;
    for (int l = 0; l < 3; ++l) {
      int N = L[l].cout, K = L[l].cin, Np = r16(N);
      gemm(cur, Wp[l], nullptr, bufY, Mpad, N, K, 0);
      k_bn_stats<<<N, 256, 0, stream>>>(bufY, M, Np, L[l].g, L[l].b, d_sc, d_sh);
      if (l < 2) {
        int ldd = r32(N);
        size_t tot = (size_t)Mpad * ldd;
        k_pack_act<<<cdiv(tot, 256), 256, 0, stream>>>(bufY, Np, M, Mpad, N,
                                                       d_sc, d_sh, 1, alt, ldd);
        half_t* t2 = cur; cur = alt; alt = t2;
      } else {
        size_t tot = (size_t)M * N;
        k_bn_relu<<<cdiv(tot, 256), 256, 0, stream>>>(bufY, Np, M, N, d_sc, d_sh);
      }
    }
  };

  // ---- SA1: 8192 pts -> 512 centers, ns=32, 3->32->32->64
  k_fps<<<BATCH, 256, 0, stream>>>(pc, 8192, 512, idx1);
  k_gather3<<<cdiv(8 * 512 * 3, 256), 256, 0, stream>>>(pc, idx1, x1, 8192, 512);
  k_ball<<<cdiv(8 * 512, 256), 256, 0, stream>>>(pc, x1, ball1, 8192, 512, 0.02f * 0.02f, 32);
  k_group<<<cdiv(8 * 512 * 32, 256), 256, 0, stream>>>(pc, nullptr, x1, ball1, bufA,
                                                       8192, 512, 32, 0, 32);
  Lay Lsa1[3] = {{f(1), f(2), f(3), 3, 32}, {f(4), f(5), f(6), 32, 32}, {f(7), f(8), f(9), 32, 64}};
  half_t* W1[3] = {packW(f(1), 32, 3), packW(f(4), 32, 32), packW(f(7), 64, 32)};
  runMLP(bufA, 131072, 131072, Lsa1, W1);
  k_max_group<<<cdiv((size_t)4096 * 64, 256), 256, 0, stream>>>(bufY, 64, 4096, 32, 64, f1);

  // ---- SA2: 512 -> 256, ns=16, 67->128->128->256
  k_fps<<<BATCH, 256, 0, stream>>>(x1, 512, 256, idx2);
  k_gather3<<<cdiv(8 * 256 * 3, 256), 256, 0, stream>>>(x1, idx2, x2, 512, 256);
  k_ball<<<cdiv(8 * 256, 256), 256, 0, stream>>>(x1, x2, ball2, 512, 256, 0.04f * 0.04f, 16);
  k_group<<<cdiv(8 * 256 * 16, 256), 256, 0, stream>>>(x1, f1, x2, ball2, bufA,
                                                       512, 256, 16, 64, 96);
  Lay Lsa2[3] = {{f(10), f(11), f(12), 67, 128}, {f(13), f(14), f(15), 128, 128},
                 {f(16), f(17), f(18), 128, 256}};
  half_t* W2[3] = {packW(f(10), 128, 67), packW(f(13), 128, 128), packW(f(16), 256, 128)};
  runMLP(bufA, 32768, 32768, Lsa2, W2);
  k_max_group<<<cdiv((size_t)2048 * 256, 256), 256, 0, stream>>>(bufY, 256, 2048, 16, 256, f2);

  // ---- SA3: 256 -> 128, ns=16, 259->256->512->512
  k_fps<<<BATCH, 256, 0, stream>>>(x2, 256, 128, idx3);
  k_gather3<<<cdiv(8 * 128 * 3, 256), 256, 0, stream>>>(x2, idx3, x3, 256, 128);
  k_ball<<<cdiv(8 * 128, 256), 256, 0, stream>>>(x2, x3, ball3, 256, 128, 0.08f * 0.08f, 16);
  k_group<<<cdiv(8 * 128 * 16, 256), 256, 0, stream>>>(x2, f2, x3, ball3, bufA,
                                                       256, 128, 16, 256, 288);
  Lay Lsa3[3] = {{f(19), f(20), f(21), 259, 256}, {f(22), f(23), f(24), 256, 512},
                 {f(25), f(26), f(27), 512, 512}};
  half_t* W3[3] = {packW(f(19), 256, 259), packW(f(22), 512, 256), packW(f(25), 512, 512)};
  runMLP(bufA, 16384, 16384, Lsa3, W3);
  k_max_group<<<cdiv((size_t)1024 * 512, 256), 256, 0, stream>>>(bufY, 512, 1024, 16, 512, f3);

  // ---- SA4 global: rows = 8*128, 515->512->1024->1024, max over 128 pts
  k_rows<<<cdiv((size_t)1024 * 544, 256), 256, 0, stream>>>(bufA, 544, 1024, 1024, x3, 3, f3, 512);
  Lay Lsa4[3] = {{f(28), f(29), f(30), 515, 512}, {f(31), f(32), f(33), 512, 1024},
                 {f(34), f(35), f(36), 1024, 1024}};
  half_t* W4[3] = {packW(f(28), 512, 515), packW(f(31), 1024, 512), packW(f(34), 1024, 1024)};
  runMLP(bufA, 1024, 1024, Lsa4, W4);
  k_max_group<<<cdiv((size_t)8 * 1024, 256), 256, 0, stream>>>(bufY, 1024, 8, 128, 1024, f4);

  // ---- SA_sep: fps(pc,4), 3->256->512->1024, max over 4 pts
  k_fps<<<BATCH, 256, 0, stream>>>(pc, 8192, 4, idxsep);
  k_gather3<<<cdiv(8 * 4 * 3, 256), 256, 0, stream>>>(pc, idxsep, xsep, 8192, 4);
  k_rows<<<cdiv((size_t)32 * 32, 256), 256, 0, stream>>>(bufA, 32, 32, 32, xsep, 3, nullptr, 0);
  Lay Lsep[3] = {{f(37), f(38), f(39), 3, 256}, {f(40), f(41), f(42), 256, 512},
                 {f(43), f(44), f(45), 512, 1024}};
  half_t* Wsep[3] = {packW(f(37), 256, 3), packW(f(40), 512, 256), packW(f(43), 1024, 512)};
  runMLP(bufA, 32, 32, Lsep, Wsep);
  k_max_group<<<cdiv((size_t)8 * 1024, 256), 256, 0, stream>>>(bufY, 1024, 8, 4, 1024, fsep);

  // ---- head: conv1(bn1,relu) -> concat sep -> conv2(bn2,relu) -> conv3
  half_t* Wc1 = packW(f(46), 1024, 2048);
  half_t* Wc2 = packW(f(50), 2048, 2048);
  half_t* Wc3 = packW(f(54), 1024, 2048);
  half_t* Wm1 = packW(f(56), 1024, 1024);
  half_t* Wm2 = packW(f(58), 1024, 1024);
  half_t* Wm3 = packW(f(60), 1152, 1024);
  half_t* Wf1 = packW(f(62), 512, 1029);
  half_t* Wf2 = packW(f(66), 512, 512);
  half_t* Wf3 = packW(f(70), 3, 512);

  k_rows<<<cdiv((size_t)16 * 2048, 256), 256, 0, stream>>>(bufA, 2048, 8, 16, f4, 1024, fsep, 1024);
  gemm(bufA, Wc1, f(47), bufY, 16, 1024, 2048, 0);
  k_bn_stats<<<1024, 256, 0, stream>>>(bufY, 8, 1024, f(48), f(49), d_sc, d_sh);
  k_bn_relu<<<cdiv((size_t)8 * 1024, 256), 256, 0, stream>>>(bufY, 1024, 8, 1024, d_sc, d_sh);

  k_rows<<<cdiv((size_t)16 * 2048, 256), 256, 0, stream>>>(bufB, 2048, 8, 16, bufY, 1024, fsep, 1024);
  gemm(bufB, Wc2, f(51), bufY, 16, 2048, 2048, 0);
  k_bn_stats<<<2048, 256, 0, stream>>>(bufY, 8, 2048, f(52), f(53), d_sc, d_sh);
  k_pack_act<<<cdiv((size_t)16 * 2048, 256), 256, 0, stream>>>(bufY, 2048, 8, 16, 2048,
                                                               d_sc, d_sh, 1, bufA, 2048);
  gemm(bufA, Wc3, f(55), gfY, 16, 1024, 2048, 0);   // gf (B x 1024)

  // ---- mlp1/2/3 -> coarse
  k_rows<<<cdiv((size_t)16 * 1024, 256), 256, 0, stream>>>(bufA, 1024, 8, 16, gfY, 1024, nullptr, 0);
  gemm(bufA, Wm1, f(57), bufY, 16, 1024, 1024, 1);
  k_rows<<<cdiv((size_t)16 * 1024, 256), 256, 0, stream>>>(bufB, 1024, 8, 16, bufY, 1024, nullptr, 0);
  gemm(bufB, Wm2, f(59), bufY, 16, 1024, 1024, 1);
  k_rows<<<cdiv((size_t)16 * 1024, 256), 256, 0, stream>>>(bufA, 1024, 8, 16, bufY, 1024, nullptr, 0);
  gemm(bufA, Wm3, f(61), czY, 16, 1152, 1024, 0);   // coarse (B x 384 x 3)

  // ---- folding decoder: 49152 rows x 1029 -> 512 -> 512 -> 3
  size_t dtot = (size_t)49152 * 1056;
  k_decfeat<<<cdiv(dtot, 256), 256, 0, stream>>>(gfY, czY, bufA);
  gemm(bufA, Wf1, f(63), bufY, 49152, 512, 1029, 0);
  k_bn_stats<<<512, 256, 0, stream>>>(bufY, 49152, 512, f(64), f(65), d_sc, d_sh);
  k_pack_act<<<cdiv((size_t)49152 * 512, 256), 256, 0, stream>>>(bufY, 512, 49152, 49152, 512,
                                                                 d_sc, d_sh, 1, bufB, 512);
  gemm(bufB, Wf2, f(67), bufY, 49152, 512, 512, 0);
  k_bn_stats<<<512, 256, 0, stream>>>(bufY, 49152, 512, f(68), f(69), d_sc, d_sh);
  k_pack_act<<<cdiv((size_t)49152 * 512, 256), 256, 0, stream>>>(bufY, 512, 49152, 49152, 512,
                                                                 d_sc, d_sh, 1, bufA, 512);
  gemm(bufA, Wf3, f(71), bufY, 49152, 3, 512, 0);   // fc3 -> (49152 x 16), cols 0..2 valid

  // ---- outputs: center | pc_out | coarse_out | fine
  k_fps<<<BATCH, 256, 0, stream>>>(pc, 8192, 128, idx128);
  float* out  = (float*)d_out;
  float* o_pc = out + BATCH * 3;
  float* o_co = o_pc + BATCH * (NPOINTS + NFINE) * 3;
  float* o_fi = o_co + BATCH * (NCOARSE + 128) * 3;
  k_center<<<1, 32, 0, stream>>>(out);
  k_fine<<<cdiv((size_t)BATCH * NFINE * 3, 256), 256, 0, stream>>>(bufY, czY, o_fi);
  k_pcout<<<cdiv((size_t)BATCH * (NPOINTS + NFINE) * 3, 256), 256, 0, stream>>>(pc, o_fi, o_pc);
  k_coarseout<<<cdiv((size_t)BATCH * (NCOARSE + 128) * 3, 256), 256, 0, stream>>>(czY, pc, idx128, o_co);
}
